// FeatureMemory_68126771249384
// MI455X (gfx1250) — matrix operand order repaired and checked by reading, not compiled, not run on hardware
//
#include <hip/hip_runtime.h>
#include <hip/hip_bf16.h>
#include <cstdint>
#include <cstddef>

// ---------------------------------------------------------------------------
// Types for CDNA5 WMMA (gfx1250, wave32)
// ---------------------------------------------------------------------------
typedef __attribute__((ext_vector_type(16))) __bf16 bf16x16;
typedef __attribute__((ext_vector_type(8)))  __bf16 bf16x8;
typedef __attribute__((ext_vector_type(8)))  float  v8f;
typedef __attribute__((ext_vector_type(4)))  float  f4;

#define WG_THREADS 256   // 8 waves of 32
#define TILE_N     32    // feats columns per workgroup (2 WMMA n-subtiles)
#define KC         64    // K elements staged per LDS stage (2 WMMA k-steps)

__device__ __forceinline__ v8f wmma_bf16(bf16x16 a, bf16x16 b, v8f c) {
  // 8 args: (neg_a, A, neg_b, B, c_mod, C, reuse_a, reuse_b)
  return __builtin_amdgcn_wmma_f32_16x16x32_bf16(false, a, false, b, (short)0, c,
                                                 false, false);
}

// Build a 16x32 bf16 A fragment for one lane.
// Lane (h = lane>>4, m = lane&15): element e -> K = e + 16h.
// `p` points at row start + (k + 16h).
__device__ __forceinline__ bf16x16 load_a_frag(const __bf16* p) {
  bf16x8 a0 = *(const bf16x8*)(p);        // K = 16h .. 16h+7
  bf16x8 a1 = *(const bf16x8*)(p + 8);    // K = 16h+8 .. 16h+15
  return __builtin_shufflevector(a0, a1, 0, 1, 2, 3, 4, 5, 6, 7,
                                 8, 9, 10, 11, 12, 13, 14, 15);
}

// ---------------------------------------------------------------------------
// Kernel 1: x -> bf16 hi/lo split, exact row norms xx, init rowmin = +inf
// one block per row of x
// ---------------------------------------------------------------------------
__global__ __launch_bounds__(WG_THREADS)
void prep_kernel(const float* __restrict__ x, __bf16* __restrict__ xhi,
                 __bf16* __restrict__ xlo, float* __restrict__ xx,
                 int* __restrict__ rowmin, int D) {
  const int row = blockIdx.x;
  const int tid = threadIdx.x;
  __shared__ float red[WG_THREADS];
  float s = 0.0f;
  const size_t base = (size_t)row * D;
  for (int k = tid; k < D; k += WG_THREADS) {
    float v = x[base + k];
    __bf16 h = (__bf16)v;
    float lo = v - (float)h;
    xhi[base + k] = h;
    xlo[base + k] = (__bf16)lo;
    s += v * v;
  }
  red[tid] = s;
  __syncthreads();
  for (int off = WG_THREADS / 2; off > 0; off >>= 1) {
    if (tid < off) red[tid] += red[tid + off];
    __syncthreads();
  }
  if (tid == 0) {
    xx[row] = red[0];
    rowmin[row] = 0x7F800000;  // +inf as int bits
  }
}

// ---------------------------------------------------------------------------
// Kernel 2: exact f32 d_ap per row: sqrt(clamp(xx + ||f_l||^2 - 2 x.f_l))
// one block per row
// ---------------------------------------------------------------------------
__global__ __launch_bounds__(WG_THREADS)
void dap_kernel(const float* __restrict__ x, const float* __restrict__ feats,
                const int* __restrict__ labels, const float* __restrict__ xx,
                float* __restrict__ dap, int D) {
  const int row = blockIdx.x;
  const int tid = threadIdx.x;
  __shared__ float rdot[WG_THREADS];
  __shared__ float rsq[WG_THREADS];
  const long long lab = labels[row];
  const size_t xb = (size_t)row * D;
  const size_t fb = (size_t)lab * D;
  float dot = 0.0f, fs = 0.0f;
  for (int k = tid; k < D; k += WG_THREADS) {
    float xv = x[xb + k];
    float fv = feats[fb + k];
    dot += xv * fv;
    fs += fv * fv;
  }
  rdot[tid] = dot;
  rsq[tid] = fs;
  __syncthreads();
  for (int off = WG_THREADS / 2; off > 0; off >>= 1) {
    if (tid < off) { rdot[tid] += rdot[tid + off]; rsq[tid] += rsq[tid + off]; }
    __syncthreads();
  }
  if (tid == 0) {
    float d2 = xx[row] + rsq[0] - 2.0f * rdot[0];
    dap[row] = sqrtf(fmaxf(d2, 1e-12f));
  }
}

// ---------------------------------------------------------------------------
// Kernel 3: fused split-bf16 WMMA GEMM + per-row min of sqrt distances.
// Each block: one 32-column feats tile x all N (<=512) rows.
// 8 waves; each wave owns 64 rows (4 WMMA M-tiles) x 2 N-subtiles.
// A-fragments for a k-step are prefetched into register arrays so the
// scheduler can issue one long global-load clause per k-step.
// ---------------------------------------------------------------------------
__global__ __launch_bounds__(WG_THREADS)
void gemm_min_kernel(const __bf16* __restrict__ xhi, const __bf16* __restrict__ xlo,
                     const float* __restrict__ feats, const int* __restrict__ labels,
                     const float* __restrict__ xx, int* __restrict__ rowmin,
                     int N, int D, long long P) {
  __shared__ float lds_xx[512];
  __shared__ int   lds_lab[512];
  __shared__ __align__(32) __bf16 Bhi[TILE_N][KC];
  __shared__ __align__(32) __bf16 Blo[TILE_N][KC];
  __shared__ float ypart[TILE_N][8];
  __shared__ float lds_yy[TILE_N];

  const int tid  = threadIdx.x;
  const int wave = tid >> 5;
  const int lane = tid & 31;
  const int mm   = lane & 15;   // A-lane row index / C column index
  const int h    = lane >> 4;   // lane half
  const int wbase = wave * 64;  // first row owned by this wave
  const long long j0 = (long long)blockIdx.x * TILE_N;

  // stage xx / labels (L2-hot, tiny)
  for (int i = tid; i < 512; i += WG_THREADS) {
    if (i < N) { lds_xx[i] = xx[i]; lds_lab[i] = labels[i]; }
    else       { lds_xx[i] = 0.0f;  lds_lab[i] = -1; }
  }

  // B staging assignment: thread -> (jj = tid>>3, 8 consecutive K slots)
  const int jj = tid >> 3;              // 0..31
  const int ko = (tid & 7) << 3;        // 0,8,...,56
  const long long jg = j0 + jj;
  const bool jvalid = (jg < P);
  const size_t frow = (size_t)(jvalid ? jg : (P - 1)) * D;

  v8f acc[4][2];
#pragma unroll
  for (int t = 0; t < 4; ++t) {
    acc[t][0] = (v8f)0.0f;
    acc[t][1] = (v8f)0.0f;
  }

  float ysq = 0.0f;

  for (int k0 = 0; k0 < D; k0 += KC) {
    __syncthreads();  // previous stage's LDS reads done
    // ---- cooperative feats tile load, f32 -> bf16 hi/lo, fused yy ----
    f4 v0 = __builtin_nontemporal_load((const f4*)(feats + frow + k0 + ko));
    f4 v1 = __builtin_nontemporal_load((const f4*)(feats + frow + k0 + ko + 4));
#pragma unroll
    for (int q = 0; q < 4; ++q) {
      float f = v0[q];
      __bf16 hb = (__bf16)f;
      Bhi[jj][ko + q] = hb;
      Blo[jj][ko + q] = (__bf16)(f - (float)hb);
      ysq += f * f;
      float g = v1[q];
      __bf16 hg = (__bf16)g;
      Bhi[jj][ko + 4 + q] = hg;
      Blo[jj][ko + 4 + q] = (__bf16)(g - (float)hg);
      ysq += g * g;
    }
    __syncthreads();

    // ---- two 16x16x32 k-steps per stage ----
#pragma unroll
    for (int ks = 0; ks < KC; ks += 32) {
      // B fragments: subtile s, lane (h, mm), element e -> K = e + 16h, col 16s+mm
      bf16x16 bh[2], bl[2];
#pragma unroll
      for (int s = 0; s < 2; ++s) {
        bh[s] = *(const bf16x16*)&Bhi[16 * s + mm][ks + 16 * h];
        bl[s] = *(const bf16x16*)&Blo[16 * s + mm][ks + 16 * h];
      }
      // prefetch all A fragments for this k-step (one long load clause)
      bf16x16 ah[4], al[4];
#pragma unroll
      for (int t = 0; t < 4; ++t) {
        int R = wbase + 16 * t + mm;
        if (R >= N) R = N - 1;  // clamped duplicate; excluded in epilogue
        const size_t ab = (size_t)R * D + (k0 + ks + 16 * h);
        ah[t] = load_a_frag(xhi + ab);
        al[t] = load_a_frag(xlo + ab);
      }
#pragma unroll
      for (int t = 0; t < 4; ++t) {
#pragma unroll
        for (int s = 0; s < 2; ++s) {
          acc[t][s] = wmma_bf16(ah[t], bh[s], acc[t][s]);  // hi*hi
          acc[t][s] = wmma_bf16(ah[t], bl[s], acc[t][s]);  // hi*lo
          acc[t][s] = wmma_bf16(al[t], bh[s], acc[t][s]);  // lo*hi
        }
      }
    }
  }

  // ---- finalize yy deterministically (fixed-order 8-way sums) ----
  ypart[jj][tid & 7] = jvalid ? ysq : 0.0f;
  __syncthreads();
  if (tid < TILE_N) {
    float s = 0.0f;
#pragma unroll
    for (int q = 0; q < 8; ++q) s += ypart[tid][q];
    lds_yy[tid] = (j0 + tid < P) ? s : __builtin_inff();
  }
  __syncthreads();

  // ---- epilogue: d = sqrt(clamp(xx + yy - 2*dot)), mask label col, row min ----
#pragma unroll
  for (int t = 0; t < 4; ++t) {
#pragma unroll
    for (int r = 0; r < 8; ++r) {
      const int R = wbase + 16 * t + 8 * h + r;
      const float xv = lds_xx[R];
      const int labR = lds_lab[R];
      float best = __builtin_inff();
#pragma unroll
      for (int s = 0; s < 2; ++s) {
        float d2 = xv + lds_yy[16 * s + mm] - 2.0f * acc[t][s][r];
        float d = sqrtf(fmaxf(d2, 1e-12f));
        if (labR == (int)(j0 + 16 * s + mm)) d = __builtin_inff();
        best = fminf(best, d);
      }
      // min across the 16 lanes holding this row's 32 columns
#pragma unroll
      for (int off = 1; off < 16; off <<= 1)
        best = fminf(best, __shfl_xor(best, off, 32));
      if (mm == 0 && R < N)
        atomicMin(rowmin + R, __float_as_int(best));  // positive floats: int order ok
    }
  }
}

// ---------------------------------------------------------------------------
// Kernel 4: loss = mean(softplus(dap - dan)), stable, deterministic reduce
// ---------------------------------------------------------------------------
__global__ __launch_bounds__(WG_THREADS)
void loss_kernel(const int* __restrict__ rowmin, const float* __restrict__ dap,
                 float* __restrict__ out, int N) {
  __shared__ float red[WG_THREADS];
  const int tid = threadIdx.x;
  float s = 0.0f;
  for (int i = tid; i < N; i += WG_THREADS) {
    float an = __int_as_float(rowmin[i]);
    float z = dap[i] - an;  // -(an - ap)
    s += fmaxf(z, 0.0f) + log1pf(expf(-fabsf(z)));
  }
  red[tid] = s;
  __syncthreads();
  for (int off = WG_THREADS / 2; off > 0; off >>= 1) {
    if (tid < off) red[tid] += red[tid + off];
    __syncthreads();
  }
  if (tid == 0) out[0] = red[0] / (float)N;
}

// ---------------------------------------------------------------------------
// Host launcher
// ---------------------------------------------------------------------------
extern "C" void kernel_launch(void* const* d_in, const int* in_sizes, int n_in,
                              void* d_out, int out_size, void* d_ws, size_t ws_size,
                              hipStream_t stream) {
  const float* x      = (const float*)d_in[0];
  const float* feats  = (const float*)d_in[1];
  const int*   labels = (const int*)d_in[2];

  const int N = in_sizes[2];                       // 512
  const int D = in_sizes[0] / N;                   // 2048
  const long long P = (long long)in_sizes[1] / D;  // 100000

  // workspace carve (all offsets 256B aligned for D=2048, N=512)
  char* w = (char*)d_ws;
  size_t off = 0;
  __bf16* xhi = (__bf16*)(w + off); off += (size_t)N * D * sizeof(__bf16);
  __bf16* xlo = (__bf16*)(w + off); off += (size_t)N * D * sizeof(__bf16);
  float*  xx  = (float*)(w + off);  off += (size_t)N * sizeof(float);
  float*  dap = (float*)(w + off);  off += (size_t)N * sizeof(float);
  int*    rowmin = (int*)(w + off); off += (size_t)N * sizeof(int);

  prep_kernel<<<N, WG_THREADS, 0, stream>>>(x, xhi, xlo, xx, rowmin, D);
  dap_kernel<<<N, WG_THREADS, 0, stream>>>(x, feats, labels, xx, dap, D);
  const int nTiles = (int)((P + TILE_N - 1) / TILE_N);
  gemm_min_kernel<<<nTiles, WG_THREADS, 0, stream>>>(xhi, xlo, feats, labels, xx,
                                                     rowmin, N, D, P);
  loss_kernel<<<1, WG_THREADS, 0, stream>>>(rowmin, dap, (float*)d_out, N);
}
